// MultiHeadSelfAttention_48902497632836
// MI455X (gfx1250) — matrix operand order repaired and checked by reading, not compiled, not run on hardware
//
#include <hip/hip_runtime.h>

// ---------------------------------------------------------------------------
// MHA forward (B=2, T=2048, D=2048, H=16, hd=128) with RoPE + causal softmax.
// bf16 WMMA for all GEMMs (f32 accum), TDM (tensor_load_to_lds) staging of the
// weight tiles with double buffering, flash-attention streaming softmax with
// transposed V. Compile-only target: gfx1250 (wave32, WGP).
// ---------------------------------------------------------------------------

#define B_   2
#define T_   2048
#define D_   2048
#define H_   16
#define HD_  128
#define SCALE_ 0.08838834764831845f   // 128^-0.5

typedef __attribute__((ext_vector_type(16))) __bf16 v16bf;
typedef __attribute__((ext_vector_type(8)))  float  v8f;
typedef __attribute__((ext_vector_type(4)))  unsigned int u32x4;
typedef __attribute__((ext_vector_type(8)))  int i32x8;
typedef __attribute__((ext_vector_type(4)))  int i32x4;

union BF16x16 {
    v16bf v;
    uint4 q[2];
};

__device__ __forceinline__ unsigned short f2bf(float f) {
    unsigned int u = __float_as_uint(f);
    unsigned int r = (u + 0x7FFFu + ((u >> 16) & 1u)) >> 16;   // RNE
    return (unsigned short)r;
}
__device__ __forceinline__ float bf2f(unsigned short h) {
    return __uint_as_float(((unsigned int)h) << 16);
}

// ---------------------------------------------------------------------------
// TDM: load a (rows x 32) bf16 tile from global (row stride K elements) into
// LDS at lds_byte_addr, padding each 64B row by 16B (LDS row stride 80B so
// ds_load_b128 fragment reads are 16B-aligned and bank-conflict-free).
// D# built per CDNA5 ISA.
// ---------------------------------------------------------------------------
__device__ __forceinline__ void tdm_load_w_tile(const unsigned short* gsrc,
                                                unsigned int lds_byte_addr,
                                                int K, int rows) {
    const unsigned long long ga = (unsigned long long)(size_t)gsrc;
    u32x4 g0;
    g0.x = 1u;                                   // count=1, user mode, no gather
    g0.y = lds_byte_addr;                        // lds_addr
    g0.z = (unsigned int)ga;                     // global_addr[31:0]
    g0.w = (unsigned int)((ga >> 32) & 0x01FFFFFFu) | (2u << 30);  // addr[56:32], type=2
    i32x8 g1;
    g1[0] = (int)((1u << 16)     // data_size = 1 (2 bytes)
                | (1u << 20)     // pad_enable
                | (3u << 22)     // pad_interval code 3 = 16 DWORDs (64B)
                | (3u << 25));   // pad_amount  code 3 =  4 DWORDs (16B)
    g1[1] = (int)(((unsigned)K & 0xFFFFu) << 16);            // tensor_dim0 lo16
    g1[2] = (int)((((unsigned)K >> 16) & 0xFFFFu)            // tensor_dim0 hi16
                | (((unsigned)rows & 0xFFFFu) << 16));       // tensor_dim1 lo16
    g1[3] = (int)(32u << 16);                                // tensor_dim1 hi=0, tile_dim0=32
    g1[4] = rows;                                            // tile_dim1
    g1[5] = K;                                               // tensor_dim0_stride lo32
    g1[6] = 0;
    g1[7] = 0;
    const i32x4 z4 = {0, 0, 0, 0};
#if __has_include(<hip/amd_detail/amd_gfx1250_TDM.h>)
    const i32x8 z8 = {0, 0, 0, 0, 0, 0, 0, 0};
    __builtin_amdgcn_tensor_load_to_lds(g0, g1, z4, z4, z8, 0);
#else
    __builtin_amdgcn_tensor_load_to_lds(g0, g1, z4, z4, 0);
#endif
}

// ---------------------------------------------------------------------------
// f32 -> bf16 conversion
// ---------------------------------------------------------------------------
__global__ __launch_bounds__(256)
void f32_to_bf16_kernel(const float* __restrict__ in, unsigned short* __restrict__ out, int n) {
    int i = blockIdx.x * 256 + threadIdx.x;
    if (i < n) out[i] = f2bf(in[i]);
}

// ---------------------------------------------------------------------------
// GEMM: Y[r,n] = sum_k A[r,k] * W[n,k]   (A: MxK bf16, W: NxK bf16 row-major)
// Block = 8 waves = 16(M) x 256(N) tile. Per k-step of 32 the 256x32 W slice
// is staged into LDS by the TDM (double buffered); each wave computes 2
// 16x16 WMMAs per step and register-pipelines its A fragment.
// Epilogue modes:
//   0: bf16 out at (b,h,t,d)          [Q,K projections]
//   1: bf16 out at (b,h,d,t)          [V projection, transposed]
//   2: f32  out row-major [r*N + n]   [final output projection]
// ---------------------------------------------------------------------------
#define WROWS 256
#define WSTRIDE_S 40     // LDS row stride in shorts: 32 data + 8 pad (80B)

__global__ __launch_bounds__(256)
void gemm_bf16_kernel(const unsigned short* __restrict__ Ab,
                      const unsigned short* __restrict__ Wb,
                      void* __restrict__ outp,
                      int M, int N, int K, int mode) {
    __shared__ unsigned short Wlds[2][WROWS * WSTRIDE_S];   // 2 x 20KB

    const int lane = threadIdx.x & 31;
    const int wid  = threadIdx.x >> 5;
    const int half = lane >> 4;     // 0: lanes 0-15, 1: lanes 16-31
    const int lcol = lane & 15;

    const int nblks = N >> 8;                 // # of 256-column blocks
    const int mtile = blockIdx.x / nblks;
    const int nblk  = blockIdx.x % nblks;

    const unsigned short* arow  = Ab + (size_t)(mtile * 16 + lcol) * K;
    const unsigned short* wtile = Wb + (size_t)(nblk * WROWS) * K;
    const unsigned int lds0 = (unsigned int)(size_t)&Wlds[0][0];
    const unsigned int lds1 = (unsigned int)(size_t)&Wlds[1][0];

    const int nk = K >> 5;
    if (wid == 0) {
        tdm_load_w_tile(wtile, lds0, K, WROWS);
        if (nk > 1) tdm_load_w_tile(wtile + 32, lds1, K, WROWS);
    }

    const v8f z8 = {0.f,0.f,0.f,0.f,0.f,0.f,0.f,0.f};
    v8f acc[2];
    acc[0] = z8; acc[1] = z8;

    // A fragment k=0 (register pipeline head)
    BF16x16 a;
    a.q[0] = *(const uint4*)(arow + half * 8);
    a.q[1] = *(const uint4*)(arow + 16 + half * 8);

    for (int i = 0; i < nk; ++i) {
        const int cur = i & 1;
        if (wid == 0) {
            if (i + 1 < nk) __builtin_amdgcn_s_wait_tensorcnt(1);
            else            __builtin_amdgcn_s_wait_tensorcnt(0);
        }
        __syncthreads();            // buffer `cur` is resident for everyone

        // pipeline: start next A fragment while WMMAs run on current one
        BF16x16 anext;
        if (i + 1 < nk) {
            const int k0n = (i + 1) << 5;
            anext.q[0] = *(const uint4*)(arow + k0n + half * 8);
            anext.q[1] = *(const uint4*)(arow + k0n + 16 + half * 8);
            __builtin_prefetch(arow + ((i + 2) << 5), 0, 1);
        }

#pragma unroll
        for (int j = 0; j < 2; ++j) {
            const int lr = wid * 32 + j * 16 + lcol;          // local W row
            const unsigned short* bp = &Wlds[cur][lr * WSTRIDE_S + half * 8];
            BF16x16 bm;
            bm.q[0] = *(const uint4*)(bp);
            bm.q[1] = *(const uint4*)(bp + 16);
            acc[j] = __builtin_amdgcn_wmma_f32_16x16x32_bf16(
                false, a.v, false, bm.v, (short)0, acc[j], false, false);
        }

        __syncthreads();            // everyone done reading buffer `cur`
        if (wid == 0 && i + 2 < nk) {
            tdm_load_w_tile(wtile + (size_t)((i + 2) << 5), cur ? lds1 : lds0, K, WROWS);
        }
        if (i + 1 < nk) a = anext;
    }

    // epilogue: C layout — VGPR jj holds row (mtile*16 + jj + 8*half), col lcol
#pragma unroll
    for (int j = 0; j < 2; ++j) {
        const int ncol = nblk * 256 + wid * 32 + j * 16 + lcol;
#pragma unroll
        for (int jj = 0; jj < 8; ++jj) {
            const int r = mtile * 16 + jj + half * 8;
            const float vv = acc[j][jj];
            if (mode == 2) {
                ((float*)outp)[(size_t)r * N + ncol] = vv;
            } else {
                const unsigned short bv = f2bf(vv);
                const int b = r / T_, t = r % T_;
                const int h = ncol >> 7, d = ncol & 127;
                if (mode == 0) {
                    ((unsigned short*)outp)[((size_t)(b * H_ + h) * T_ + t) * HD_ + d] = bv;
                } else {
                    ((unsigned short*)outp)[((size_t)(b * H_ + h) * HD_ + d) * T_ + t] = bv;
                }
            }
        }
    }
}

// ---------------------------------------------------------------------------
// RoPE applied in-place to Q,K (bf16, layout (b,h,t,hd)); f32 math.
// ---------------------------------------------------------------------------
__global__ __launch_bounds__(256)
void rope_kernel(unsigned short* __restrict__ Qb, unsigned short* __restrict__ Kb) {
    const int idx = blockIdx.x * 256 + threadIdx.x;     // over B*H*T*64
    const int d  = idx & 63;
    const int t  = (idx >> 6) & (T_ - 1);
    const int bh = idx >> 17;
    if (bh >= B_ * H_) return;

    const float inv = __expf(-9.210340371976184f * (float)d * (1.0f / 64.0f)); // theta^(-d/64)
    const float ang = (float)t * inv;
    const float c = __cosf(ang), s = __sinf(ang);

    const size_t base = ((size_t)bh * T_ + t) * HD_ + d;
    {
        const float q1 = bf2f(Qb[base]), q2 = bf2f(Qb[base + 64]);
        Qb[base]      = f2bf(q1 * c - q2 * s);
        Qb[base + 64] = f2bf(q2 * c + q1 * s);
    }
    {
        const float k1 = bf2f(Kb[base]), k2 = bf2f(Kb[base + 64]);
        Kb[base]      = f2bf(k1 * c - k2 * s);
        Kb[base + 64] = f2bf(k2 * c + k1 * s);
    }
}

// ---------------------------------------------------------------------------
// Flash attention. Grid: (T/128, B*H), block 256 (8 waves). Each wave owns a
// 16-query tile, streams causal key tiles of 32. Q/K in (b,h,t,hd) bf16,
// V transposed (b,h,hd,t) bf16. All fragment loads for a WMMA group are
// issued before the WMMA chain so loads overlap matrix math.
// ---------------------------------------------------------------------------
__global__ __launch_bounds__(256)
void flash_attn_kernel(const unsigned short* __restrict__ Qb,
                       const unsigned short* __restrict__ Kb,
                       const unsigned short* __restrict__ Vtb,
                       unsigned short* __restrict__ Ob) {
    __shared__ unsigned short Plds[8][16 * 32];   // per-wave 16x32 bf16 P tile

    const int lane = threadIdx.x & 31;
    const int wid  = threadIdx.x >> 5;
    const int half = lane >> 4;
    const int lcol = lane & 15;
    const int bh   = blockIdx.y;
    const int q0   = blockIdx.x * 128 + wid * 16;

    // Preload Q A-fragments: 4 chunks of K=32 covering head_dim=128
    BF16x16 qa[4];
    const unsigned short* qrow = Qb + ((size_t)bh * T_ + q0 + lcol) * HD_;
#pragma unroll
    for (int c = 0; c < 4; ++c) {
        qa[c].q[0] = *(const uint4*)(qrow + c * 32 + half * 8);
        qa[c].q[1] = *(const uint4*)(qrow + c * 32 + 16 + half * 8);
    }

    const v8f z8 = {0.f,0.f,0.f,0.f,0.f,0.f,0.f,0.f};
    v8f acc[8];
#pragma unroll
    for (int nt = 0; nt < 8; ++nt) acc[nt] = z8;

    float mrun[8], lrun[8];
#pragma unroll
    for (int jj = 0; jj < 8; ++jj) { mrun[jj] = -1e30f; lrun[jj] = 0.f; }

    for (int kt0 = 0; kt0 < q0 + 16; kt0 += 32) {
        // ---- load all K fragments for this 32-key step ----
        const unsigned short* kbase = Kb + ((size_t)bh * T_ + kt0 + lcol) * HD_;
        BF16x16 kb0[4], kb1[4];
#pragma unroll
        for (int c = 0; c < 4; ++c) {
            const unsigned short* p0 = kbase + c * 32 + half * 8;
            kb0[c].q[0] = *(const uint4*)(p0);
            kb0[c].q[1] = *(const uint4*)(p0 + 16);
            const unsigned short* p1 = p0 + (size_t)16 * HD_;
            kb1[c].q[0] = *(const uint4*)(p1);
            kb1[c].q[1] = *(const uint4*)(p1 + 16);
        }
        // ---- scores: two independent WMMA chains ----
        v8f s0 = z8, s1 = z8;
#pragma unroll
        for (int c = 0; c < 4; ++c) {
            s0 = __builtin_amdgcn_wmma_f32_16x16x32_bf16(
                false, qa[c].v, false, kb0[c].v, (short)0, s0, false, false);
            s1 = __builtin_amdgcn_wmma_f32_16x16x32_bf16(
                false, qa[c].v, false, kb1[c].v, (short)0, s1, false, false);
        }

        // ---- causal mask + online softmax (rows live in 16-lane halves) ----
        float pscale[8];
#pragma unroll
        for (int jj = 0; jj < 8; ++jj) {
            const int qrowi = q0 + jj + half * 8;
            float a  = (kt0 + lcol      <= qrowi) ? s0[jj] * SCALE_ : -1e30f;
            float b2 = (kt0 + 16 + lcol <= qrowi) ? s1[jj] * SCALE_ : -1e30f;

            float mloc = fmaxf(a, b2);
#pragma unroll
            for (int off = 8; off >= 1; off >>= 1)
                mloc = fmaxf(mloc, __shfl_xor(mloc, off, 32));

            const float mnew = fmaxf(mrun[jj], mloc);
            const float sc   = __expf(mrun[jj] - mnew);
            const float p0   = __expf(a - mnew);
            const float p1   = __expf(b2 - mnew);

            float ps = p0 + p1;
#pragma unroll
            for (int off = 8; off >= 1; off >>= 1)
                ps += __shfl_xor(ps, off, 32);

            lrun[jj] = lrun[jj] * sc + ps;
            mrun[jj] = mnew;
            pscale[jj] = sc;

            // stash P (bf16) in C-layout: row m = jj + 8*half, cols lcol / lcol+16
            const int m = jj + half * 8;
            Plds[wid][m * 32 + lcol]      = f2bf(p0);
            Plds[wid][m * 32 + 16 + lcol] = f2bf(p1);
        }

        // ---- load all V fragments (contiguous thanks to transposed V) ----
        BF16x16 vb[8];
#pragma unroll
        for (int nt = 0; nt < 8; ++nt) {
            const unsigned short* vbase =
                Vtb + ((size_t)bh * HD_ + nt * 16 + lcol) * T_ + kt0 + half * 8;
            vb[nt].q[0] = *(const uint4*)(vbase);
            vb[nt].q[1] = *(const uint4*)(vbase + 16);
        }

        // rescale running accumulators while loads are in flight
#pragma unroll
        for (int nt = 0; nt < 8; ++nt)
#pragma unroll
            for (int jj = 0; jj < 8; ++jj) acc[nt][jj] *= pscale[jj];

        // cross-lane LDS handoff: wait for all P stores from this wave
        asm volatile("s_wait_dscnt 0x0" ::: "memory");

        // reload P in A-layout (16x32): lane row = lcol, K split by half
        BF16x16 pa;
        pa.q[0] = *(const uint4*)&Plds[wid][lcol * 32 + half * 8];
        pa.q[1] = *(const uint4*)&Plds[wid][lcol * 32 + 16 + half * 8];

        // ---- PV: 8 independent WMMA chains ----
#pragma unroll
        for (int nt = 0; nt < 8; ++nt) {
            acc[nt] = __builtin_amdgcn_wmma_f32_16x16x32_bf16(
                false, pa.v, false, vb[nt].v, (short)0, acc[nt], false, false);
        }
    }

    // ---- normalize and write O as bf16 row-major (b*t, D) ----
    const int b = bh / H_;
    const int h = bh % H_;
#pragma unroll
    for (int nt = 0; nt < 8; ++nt) {
#pragma unroll
        for (int jj = 0; jj < 8; ++jj) {
            const int m = jj + half * 8;
            const float o = acc[nt][jj] / lrun[jj];
            Ob[((size_t)(b * T_ + q0 + m)) * D_ + h * HD_ + nt * 16 + lcol] = f2bf(o);
        }
    }
}

// ---------------------------------------------------------------------------
// Host-side launcher
// ---------------------------------------------------------------------------
extern "C" void kernel_launch(void* const* d_in, const int* in_sizes, int n_in,
                              void* d_out, int out_size, void* d_ws, size_t ws_size,
                              hipStream_t stream) {
    (void)in_sizes; (void)n_in; (void)out_size; (void)ws_size;

    const float* x  = (const float*)d_in[0];
    const float* Wq = (const float*)d_in[1];
    const float* Wk = (const float*)d_in[2];
    const float* Wv = (const float*)d_in[3];
    const float* Wo = (const float*)d_in[4];
    float* out = (float*)d_out;

    const size_t BT = (size_t)B_ * T_;          // 4096 rows
    const size_t XE = BT * D_;                  // 8.39M elements
    const size_t WE = (size_t)D_ * D_;          // 4.19M elements

    char* ws = (char*)d_ws;
    size_t off = 0;
    unsigned short* Wbuf = (unsigned short*)(ws + off); off += WE * 2;   // reused weight slot
    unsigned short* xb   = (unsigned short*)(ws + off); off += XE * 2;   // x bf16; reused as Ob
    unsigned short* Qb   = (unsigned short*)(ws + off); off += XE * 2;   // (b,h,t,hd)
    unsigned short* Kb   = (unsigned short*)(ws + off); off += XE * 2;   // (b,h,t,hd)
    unsigned short* Vtb  = (unsigned short*)(ws + off); off += XE * 2;   // (b,h,hd,t)
    unsigned short* Ob   = xb;                                           // alias: x dead after V

    const int cvtX = (int)((XE + 255) / 256);
    const int cvtW = (int)((WE + 255) / 256);
    const int gemmBlocks = (int)((BT / 16) * (D_ / 256));                 // 2048

    // x -> bf16
    f32_to_bf16_kernel<<<cvtX, 256, 0, stream>>>(x, xb, (int)XE);

    // Q = x @ Wq^T  -> (b,h,t,hd)
    f32_to_bf16_kernel<<<cvtW, 256, 0, stream>>>(Wq, Wbuf, (int)WE);
    gemm_bf16_kernel<<<gemmBlocks, 256, 0, stream>>>(xb, Wbuf, Qb, (int)BT, D_, D_, 0);

    // K = x @ Wk^T  -> (b,h,t,hd)
    f32_to_bf16_kernel<<<cvtW, 256, 0, stream>>>(Wk, Wbuf, (int)WE);
    gemm_bf16_kernel<<<gemmBlocks, 256, 0, stream>>>(xb, Wbuf, Kb, (int)BT, D_, D_, 0);

    // V = x @ Wv^T  -> transposed (b,h,hd,t)
    f32_to_bf16_kernel<<<cvtW, 256, 0, stream>>>(Wv, Wbuf, (int)WE);
    gemm_bf16_kernel<<<gemmBlocks, 256, 0, stream>>>(xb, Wbuf, Vtb, (int)BT, D_, D_, 1);

    // RoPE on Q,K
    const int ropeBlocks = (B_ * H_ * T_ * 64) / 256;                     // 16384
    rope_kernel<<<ropeBlocks, 256, 0, stream>>>(Qb, Kb);

    // Flash attention -> Ob (bf16 row-major (b*t, D)); overwrites xb (x is dead)
    flash_attn_kernel<<<dim3(T_ / 128, B_ * H_), 256, 0, stream>>>(Qb, Kb, Vtb, Ob);

    // out = O @ Wo^T  (f32)
    f32_to_bf16_kernel<<<cvtW, 256, 0, stream>>>(Wo, Wbuf, (int)WE);
    gemm_bf16_kernel<<<gemmBlocks, 256, 0, stream>>>(Ob, Wbuf, out, (int)BT, D_, D_, 2);
}
